// AttentionLayer_32933809226388
// MI455X (gfx1250) — compile-verified, matrix operand order checked by reading, not run on hardware
//
#include <hip/hip_runtime.h>

typedef __attribute__((ext_vector_type(8)))  float   v8f;
typedef __attribute__((ext_vector_type(16))) __bf16  v16bf;
typedef __attribute__((ext_vector_type(8)))  __bf16  v8bf;
typedef __attribute__((ext_vector_type(4)))  int     v4i;

#define B_   2
#define S_   2048
#define DM   2048
#define H_   16
#define NCAT 6144
#define QS_OFF 0
#define KS_OFF 1024
#define QG_OFF 2048
#define KG_OFF 3072
#define V_OFF  4096

// ---------- async global->LDS staging (CDNA5 GLOBAL_LOAD_ASYNC_TO_LDS_B128) ----------
#if __has_builtin(__builtin_amdgcn_global_load_async_to_lds_b128)
#define ASYNC_LDS 1
#else
#define ASYNC_LDS 0
#endif

#if ASYNC_LDS
__device__ __forceinline__ void async_cp16(const void* g, void* l) {
    // global aperture: flat address == addrspace(1) address.
    // LDS aperture: flat address bits [31:0] are the LDS byte offset.
    __builtin_amdgcn_global_load_async_to_lds_b128(
        (__attribute__((address_space(1))) v4i*)(unsigned long long)g,
        (__attribute__((address_space(3))) v4i*)(unsigned int)(unsigned long long)l,
        0, 0);
}
#endif

__device__ __forceinline__ void async_wait() {
#if ASYNC_LDS
    asm volatile("s_wait_asynccnt 0x0" ::: "memory");
#endif
}

// ---------- WMMA helpers ----------
// 16-bit A/B fragment: lane holds row (lane%16); two 16B chunks at
// K offsets {g*8, 16+g*8} where g = lane/16 (per CDNA5 ISA 7.12.2).
__device__ __forceinline__ v16bf ld_frag(const __bf16* p) {
    union { v16bf v; v8bf h[2]; } u;
    u.h[0] = *(const v8bf*)(p);
    u.h[1] = *(const v8bf*)(p + 16);
    return u.v;
}

__device__ __forceinline__ v8f wmma_bf16(v16bf a, v16bf b, v8f c) {
    return __builtin_amdgcn_wmma_f32_16x16x32_bf16(
        false, a, false, b, (short)0, c, false, false);
}

// ---------- elementwise kernels ----------
__global__ void k_cast(const float* __restrict__ src, __bf16* __restrict__ dst, int n) {
    int i = blockIdx.x * blockDim.x + threadIdx.x;
    if (i < n) dst[i] = (__bf16)src[i];
}

__global__ void k_cast_cols(const float* __restrict__ src, __bf16* __restrict__ dst,
                            int ncols, int ldDst, int colOff, int n) {
    int i = blockIdx.x * blockDim.x + threadIdx.x;
    if (i < n) {
        int r = i / ncols;
        int c = i - r * ncols;
        dst[(long)r * ldDst + colOff + c] = (__bf16)src[i];
    }
}

// In-place RoPE on the geo blocks of Y (bf16). One thread per (b,s,h,d<32),
// handles both q_geo and k_geo, both half-pairs.
__global__ void k_rope(__bf16* __restrict__ Y) {
    int i = blockIdx.x * blockDim.x + threadIdx.x;
    if (i >= B_ * S_ * H_ * 32) return;
    int d = i & 31;
    int h = (i >> 5) & 15;
    int s = (i >> 9) & 2047;
    int b = i >> 20;
    // inv_freq = 10000^(-d/32) = exp(-d * ln(10000)/32)
    float invf = __expf(-(float)d * 0.28782313662425572f);
    float ang  = (float)s * invf;
    float sn, cs;
    __sincosf(ang, &sn, &cs);
    long rowbase = (long)(b * S_ + s) * NCAT + h * 64;
    #pragma unroll
    for (int k = 0; k < 2; ++k) {
        __bf16* p = Y + rowbase + (k == 0 ? QG_OFF : KG_OFF);
        float x1 = (float)p[d];
        float x2 = (float)p[d + 32];
        p[d]      = (__bf16)(x1 * cs - x2 * sn);
        p[d + 32] = (__bf16)(x1 * sn + x2 * cs);
    }
}

// ---------- generic bf16 WMMA GEMM: C(MxN) = A(MxK) @ B(KxN) ----------
// 256 threads = 8 waves. Block tile 128x128, K-step 32.
// Wave (wm=wave>>1, wn=wave&1) computes 32x64 -> 2x4 WMMA tiles.
template <bool OUT_BF16>
__global__ __launch_bounds__(256)
void gemm_bf16_kernel(const __bf16* __restrict__ A, const __bf16* __restrict__ Bm,
                      void* __restrict__ Cout, int M, int N, int K) {
    __shared__ __align__(16) __bf16 As[128 * 40];       // row-major, pad 8
    __shared__ __align__(16) __bf16 Bs[128 * 40];       // transposed: Bs[n][k]
    const int tid  = threadIdx.x;
    const int wave = tid >> 5, lane = tid & 31;
    const int wm = wave >> 1, wn = wave & 1;
    const int lm = lane & 15, lg = lane >> 4;
    const int m0 = blockIdx.y * 128, n0 = blockIdx.x * 128;

    v8f acc[2][4];
    #pragma unroll
    for (int i = 0; i < 2; ++i)
        #pragma unroll
        for (int j = 0; j < 4; ++j) acc[i][j] = {};

    for (int k0 = 0; k0 < K; k0 += 32) {
        // Stage A: 128x32 = 512 x 8-elem chunks, 2 per thread (async -> LDS).
        #pragma unroll
        for (int i = 0; i < 2; ++i) {
            int c   = tid * 2 + i;
            int row = c >> 2, kc = c & 3;
            const __bf16* g = A + (long)(m0 + row) * K + k0 + kc * 8;
            __bf16* l = As + row * 40 + kc * 8;
#if ASYNC_LDS
            async_cp16(g, l);
#else
            *(v8bf*)l = *(const v8bf*)g;
#endif
        }
        // Stage B transposed: 32(k) x 128(n), 512 chunks, 2 per thread.
        #pragma unroll
        for (int i = 0; i < 2; ++i) {
            int c  = tid * 2 + i;
            int kr = c >> 4, nc = c & 15;
            v8bf val = *(const v8bf*)(Bm + (long)(k0 + kr) * N + n0 + nc * 8);
            #pragma unroll
            for (int j = 0; j < 8; ++j) Bs[(nc * 8 + j) * 40 + kr] = val[j];
        }
        async_wait();
        __syncthreads();

        v16bf af[2], bfr[4];
        #pragma unroll
        for (int mt = 0; mt < 2; ++mt)
            af[mt] = ld_frag(As + (wm * 32 + mt * 16 + lm) * 40 + lg * 8);
        #pragma unroll
        for (int nt = 0; nt < 4; ++nt)
            bfr[nt] = ld_frag(Bs + (wn * 64 + nt * 16 + lm) * 40 + lg * 8);
        #pragma unroll
        for (int mt = 0; mt < 2; ++mt)
            #pragma unroll
            for (int nt = 0; nt < 4; ++nt)
                acc[mt][nt] = wmma_bf16(af[mt], bfr[nt], acc[mt][nt]);
        __syncthreads();
    }

    // Epilogue: C/D layout -> lane (n%16 + 16*(m/8)), reg m%8.
    #pragma unroll
    for (int mt = 0; mt < 2; ++mt)
        #pragma unroll
        for (int nt = 0; nt < 4; ++nt)
            #pragma unroll
            for (int r = 0; r < 8; ++r) {
                int m = m0 + wm * 32 + mt * 16 + lg * 8 + r;
                int n = n0 + wn * 64 + nt * 16 + lm;
                if (OUT_BF16)
                    ((__bf16*)Cout)[(long)m * N + n] = (__bf16)acc[mt][nt][r];
                else
                    ((float*)Cout)[(long)m * N + n] = acc[mt][nt][r];
            }
}

// ---------- flash attention ----------
// Grid: (S/64, B*H). 128 threads = 4 waves; wave w owns queries q0+16w..+15.
__global__ __launch_bounds__(128)
void flash_kernel(const __bf16* __restrict__ Y, const float* __restrict__ gate,
                  __bf16* __restrict__ O) {
    __shared__ __align__(16) __bf16 Ks[32 * 72];      // key-major, pad to 72
    __shared__ __align__(16) __bf16 Kg[32 * 72];
    __shared__ __align__(16) __bf16 Vt[128 * 40];     // transposed V: [d][key]
    __shared__ __align__(16) __bf16 Pl[4][16 * 40];   // per-wave P scratch

    const int tid  = threadIdx.x;
    const int wave = tid >> 5, lane = tid & 31;
    const int lm = lane & 15, lg = lane >> 4;
    const int bh = blockIdx.y, b = bh / H_, h = bh % H_;
    const int q0 = blockIdx.x * 64;
    const int qw = q0 + wave * 16;

    // gate: scores = (1/8)*(g*sem + (1-g)*geo)
    float gvl = gate[h];
    float gs  = (1.0f / (1.0f + __expf(-gvl))) * 0.125f;
    float gg  = 0.125f - gs;

    // Q fragments (loop-invariant): two K-chunks each for sem and geo.
    const long rowQ = (long)(b * S_ + qw + lm) * NCAT + h * 64;
    v16bf qs[2], qg[2];
    #pragma unroll
    for (int kc = 0; kc < 2; ++kc) {
        qs[kc] = ld_frag(Y + rowQ + QS_OFF + kc * 32 + lg * 8);
        qg[kc] = ld_frag(Y + rowQ + QG_OFF + kc * 32 + lg * 8);
    }

    float mrow[8], lrow[8];
    v8f o[8];
    #pragma unroll
    for (int r = 0; r < 8; ++r) { mrow[r] = -1e30f; lrow[r] = 0.0f; }
    #pragma unroll
    for (int db = 0; db < 8; ++db) o[db] = {};

    const int nkv = q0 / 32 + 2;
    for (int t = 0; t < nkv; ++t) {
        const int kv0 = t * 32;
        // ---- stage K_sem, K_geo (32x64 each, async) and V transposed (128x32) ----
        #pragma unroll
        for (int i = 0; i < 2; ++i) {
            int c = tid * 2 + i;               // 0..255
            int row = c >> 3, kc = c & 7;
            long base = (long)(b * S_ + kv0 + row) * NCAT + h * 64 + kc * 8;
#if ASYNC_LDS
            async_cp16(Y + base + KS_OFF, Ks + row * 72 + kc * 8);
            async_cp16(Y + base + KG_OFF, Kg + row * 72 + kc * 8);
#else
            *(v8bf*)(Ks + row * 72 + kc * 8) = *(const v8bf*)(Y + base + KS_OFF);
            *(v8bf*)(Kg + row * 72 + kc * 8) = *(const v8bf*)(Y + base + KG_OFF);
#endif
        }
        #pragma unroll
        for (int i = 0; i < 4; ++i) {
            int c = tid * 4 + i;               // 0..511
            int key = c >> 4, dc = c & 15;
            v8bf val = *(const v8bf*)(Y + (long)(b * S_ + kv0 + key) * NCAT
                                        + V_OFF + h * 128 + dc * 8);
            #pragma unroll
            for (int j = 0; j < 8; ++j) Vt[(dc * 8 + j) * 40 + key] = val[j];
        }
        async_wait();
        __syncthreads();

        if (kv0 <= qw + 15) {                 // wave has unmasked work
            // ---- scores: two 16-key column blocks, K-dim 64 = 2 WMMAs each ----
            v8f ss[2], sg[2];
            ss[0] = {}; ss[1] = {}; sg[0] = {}; sg[1] = {};
            #pragma unroll
            for (int blk = 0; blk < 2; ++blk)
                #pragma unroll
                for (int kc = 0; kc < 2; ++kc) {
                    v16bf kb = ld_frag(Ks + (blk * 16 + lm) * 72 + kc * 32 + lg * 8);
                    ss[blk] = wmma_bf16(qs[kc], kb, ss[blk]);
                    v16bf kbg = ld_frag(Kg + (blk * 16 + lm) * 72 + kc * 32 + lg * 8);
                    sg[blk] = wmma_bf16(qg[kc], kbg, sg[blk]);
                }
            // ---- gate mix + causal mask ----
            float sc[2][8];
            const bool need_mask = (kv0 + 31 > qw);
            #pragma unroll
            for (int blk = 0; blk < 2; ++blk)
                #pragma unroll
                for (int r = 0; r < 8; ++r) {
                    float s = gs * ss[blk][r] + gg * sg[blk][r];
                    if (need_mask) {
                        int key = kv0 + blk * 16 + lm;
                        int qq  = qw + lg * 8 + r;
                        if (key > qq) s = -1e30f;
                    }
                    sc[blk][r] = s;
                }
            // ---- online softmax (row reductions across 16 lanes of half) ----
            float alpha[8];
            #pragma unroll
            for (int r = 0; r < 8; ++r) {
                float mx = fmaxf(sc[0][r], sc[1][r]);
                #pragma unroll
                for (int d = 1; d < 16; d <<= 1)
                    mx = fmaxf(mx, __shfl_xor(mx, d, 32));
                float mnew = fmaxf(mrow[r], mx);
                alpha[r] = __expf(mrow[r] - mnew);
                mrow[r] = mnew;
            }
            #pragma unroll
            for (int r = 0; r < 8; ++r) {
                float p0 = __expf(sc[0][r] - mrow[r]);
                float p1 = __expf(sc[1][r] - mrow[r]);
                float sum = p0 + p1;
                #pragma unroll
                for (int d = 1; d < 16; d <<= 1)
                    sum += __shfl_xor(sum, d, 32);
                lrow[r] = lrow[r] * alpha[r] + sum;
                int m = lg * 8 + r;           // C-layout row -> P row
                Pl[wave][m * 40 + lm]      = (__bf16)p0;
                Pl[wave][m * 40 + 16 + lm] = (__bf16)p1;
            }
            __asm__ volatile("s_wait_dscnt 0" ::: "memory");
            // P as A-fragment (16q x 32key)
            v16bf pf = ld_frag(Pl[wave] + lm * 40 + lg * 8);
            // ---- O += P @ V : 8 x (16x16) output tiles over DV=128 ----
            #pragma unroll
            for (int db = 0; db < 8; ++db) {
                v16bf vb = ld_frag(Vt + (db * 16 + lm) * 40 + lg * 8);
                #pragma unroll
                for (int r = 0; r < 8; ++r) o[db][r] *= alpha[r];
                o[db] = wmma_bf16(pf, vb, o[db]);
            }
        }
        __syncthreads();
    }

    // ---- epilogue: normalize and write attn_out bf16 [b*S+q][h*128+d] ----
    float rl[8];
    #pragma unroll
    for (int r = 0; r < 8; ++r) rl[r] = (lrow[r] > 0.0f) ? 1.0f / lrow[r] : 0.0f;
    #pragma unroll
    for (int db = 0; db < 8; ++db)
        #pragma unroll
        for (int r = 0; r < 8; ++r) {
            long row = (long)(b * S_ + qw + lg * 8 + r);
            O[row * (H_ * 128) + h * 128 + db * 16 + lm] = (__bf16)(o[db][r] * rl[r]);
        }
}

// ---------- host launch ----------
extern "C" void kernel_launch(void* const* d_in, const int* in_sizes, int n_in,
                              void* d_out, int out_size, void* d_ws, size_t ws_size,
                              hipStream_t stream) {
    (void)in_sizes; (void)n_in; (void)out_size; (void)ws_size;
    const float* x      = (const float*)d_in[0];
    const float* wq_sem = (const float*)d_in[1];
    const float* wk_sem = (const float*)d_in[2];
    const float* wq_geo = (const float*)d_in[3];
    const float* wk_geo = (const float*)d_in[4];
    const float* wv     = (const float*)d_in[5];
    const float* wo     = (const float*)d_in[6];
    const float* gate   = (const float*)d_in[7];

    // workspace carve (all offsets 16B aligned)
    __bf16* xbf  = (__bf16*)d_ws;                      // 4096 x 2048
    __bf16* wcat = xbf  + (size_t)B_ * S_ * DM;        // 2048 x 6144
    __bf16* wobf = wcat + (size_t)DM * NCAT;           // 2048 x 2048
    __bf16* ybf  = wobf + (size_t)DM * DM;             // 4096 x 6144
    __bf16* attn = ybf  + (size_t)B_ * S_ * NCAT;      // 4096 x 2048

    const int TB = 256;
    int nX = B_ * S_ * DM;
    k_cast<<<(nX + TB - 1) / TB, TB, 0, stream>>>(x, xbf, nX);
    int nW1 = DM * 1024;
    k_cast_cols<<<(nW1 + TB - 1) / TB, TB, 0, stream>>>(wq_sem, wcat, 1024, NCAT, QS_OFF, nW1);
    k_cast_cols<<<(nW1 + TB - 1) / TB, TB, 0, stream>>>(wk_sem, wcat, 1024, NCAT, KS_OFF, nW1);
    k_cast_cols<<<(nW1 + TB - 1) / TB, TB, 0, stream>>>(wq_geo, wcat, 1024, NCAT, QG_OFF, nW1);
    k_cast_cols<<<(nW1 + TB - 1) / TB, TB, 0, stream>>>(wk_geo, wcat, 1024, NCAT, KG_OFF, nW1);
    int nWV = DM * 2048;
    k_cast_cols<<<(nWV + TB - 1) / TB, TB, 0, stream>>>(wv, wcat, 2048, NCAT, V_OFF, nWV);
    int nWo = DM * DM;
    k_cast<<<(nWo + TB - 1) / TB, TB, 0, stream>>>(wo, wobf, nWo);

    // fused QKV projection: Y(4096x6144) = xbf @ wcat  (bf16 out)
    gemm_bf16_kernel<true><<<dim3(NCAT / 128, (B_ * S_) / 128), 256, 0, stream>>>(
        xbf, wcat, (void*)ybf, B_ * S_, NCAT, DM);

    int nR = B_ * S_ * H_ * 32;
    k_rope<<<(nR + TB - 1) / TB, TB, 0, stream>>>(ybf);

    flash_kernel<<<dim3(S_ / 64, B_ * H_), 128, 0, stream>>>(ybf, gate, attn);

    // out(4096x2048, f32) = attn @ wobf
    gemm_bf16_kernel<false><<<dim3(DM / 128, (B_ * S_) / 128), 256, 0, stream>>>(
        attn, wobf, d_out, B_ * S_, DM, DM);
}
